// H_H_EdgeApplyModule_63900523430461
// MI455X (gfx1250) — compile-verified, matrix operand order checked by reading, not run on hardware
//
#include <hip/hip_runtime.h>
#include <stdint.h>
#include <stddef.h>

// ---------------------------------------------------------------------------
// Types for WMMA fragments (CDNA5 / gfx1250, wave32)
// ---------------------------------------------------------------------------
typedef __attribute__((ext_vector_type(16))) __bf16       v16bf;
typedef __attribute__((ext_vector_type(2)))  __bf16       v2bf;
typedef __attribute__((ext_vector_type(8)))  float        v8f;
typedef __attribute__((ext_vector_type(4)))  unsigned int u32x4;
typedef __attribute__((ext_vector_type(4)))  float        f32x4;

// Explicit global-address-space pointers: keep gathered loads on the
// GLOBAL_LOAD path (pointer selects otherwise defeat AS inference -> FLAT).
typedef const float __attribute__((address_space(1)))*          gfp;
typedef const f32x4 __attribute__((address_space(1)))*          gf4p;
typedef const unsigned short __attribute__((address_space(1)))* gu16p;
typedef const u32x4 __attribute__((address_space(1)))*          gu4p;

union FragAB {
  v16bf v;
  u32x4 u[2];
};

__device__ __forceinline__ unsigned int pack_bf16(float lo, float hi) {
  v2bf t;
  t.x = (__bf16)lo;
  t.y = (__bf16)hi;
  return __builtin_bit_cast(unsigned int, t);  // v_cvt_pk_bf16_f32
}

__device__ __forceinline__ u32x4 pack8(f32x4 f0, f32x4 f1) {
  u32x4 st;
  st.x = pack_bf16(f0.x, f0.y);
  st.y = pack_bf16(f0.z, f0.w);
  st.z = pack_bf16(f1.x, f1.y);
  st.w = pack_bf16(f1.z, f1.w);
  return st;
}

#define LDA 40  // LDS row stride in bf16 elements (32 data + 8 pad, 80B, 16B aligned)
#define ABUF (64 * LDA)

// ---------------------------------------------------------------------------
// Weight prep: fp32 [K][256] -> bf16 [256][K] (transposed, K-contig, padded)
// ---------------------------------------------------------------------------
__global__ void prep_w_main(const float* __restrict__ W, unsigned short* __restrict__ Wt) {
  const int k = blockIdx.x;   // 0..3071
  const int n = threadIdx.x;  // 0..255
  Wt[(size_t)n * 3072 + k] =
      __builtin_bit_cast(unsigned short, (__bf16)W[(size_t)k * 256 + n]);
}

__global__ void prep_w_lang(const float* __restrict__ W, unsigned short* __restrict__ Wt) {
  const int k = blockIdx.x;   // 0..607 (600 real + 8 zero pad)
  const int n = threadIdx.x;  // 0..255
  float v = (k < 600) ? W[(size_t)k * 256 + n] : 0.0f;
  Wt[(size_t)n * 608 + k] = __builtin_bit_cast(unsigned short, (__bf16)v);
}

// ---------------------------------------------------------------------------
// GEMM 1: e_f = relu([n_f[src] | s_f | n_f[dst]] @ W1 + b1)
//   M tile = 64 edges per workgroup, N = 256 (8 waves x 32), K = 3072 (96x32)
//   Double-buffered LDS A tile; one barrier per K-step.
// ---------------------------------------------------------------------------
__global__ __launch_bounds__(256) void edge_fc_main(
    const float* __restrict__ n_f,
    const float* __restrict__ s_f,
    const unsigned short* __restrict__ Wt,  // bf16 [256][3072]
    const float* __restrict__ bias,
    const int* __restrict__ src,
    const int* __restrict__ dst,
    float* __restrict__ out) {
  __shared__ __align__(128) unsigned short As[2 * ABUF];

  const int tid  = threadIdx.x;
  const int lane = tid & 31;
  const int wave = tid >> 5;       // 0..7 -> N slice
  const int hsel = lane >> 4;      // 0 or 1 (lanes 0-15 vs 16-31)
  const int lm   = lane & 15;
  const long tileM = (long)blockIdx.x * 64;

  // A-staging assignment: 4 threads per row, 8 floats each
  const int srow  = tid >> 2;      // 0..63
  const int spart = tid & 3;       // 0..3
  const long e = tileM + srow;
  gfp pA = (gfp)(n_f + (long)src[e] * 1024 + spart * 8);
  gfp pB = (gfp)(s_f + e * 1024 + spart * 8);
  gfp pC = (gfp)(n_f + (long)dst[e] * 1024 + spart * 8);

  // Branch-free gather pointer for K-tile kb (kb -> segment is uniform).
  auto stage_ptr = [&](int kb) -> gfp {
    const int seg = kb >> 5;             // 0: n_f[src], 1: s_f, 2: n_f[dst]
    gfp p = pA;
    p = (seg == 1) ? pB : p;
    p = (seg == 2) ? pC : p;
    return p + (kb & 31) * 32;
  };

  v8f acc[4][2] = {};

  // Prologue: stage K-tile 0 into buffer 0
  {
    gfp q = stage_ptr(0);
    f32x4 f0 = *(gf4p)(q);
    f32x4 f1 = *(gf4p)(q + 4);
    *(u32x4*)(&As[srow * LDA + spart * 8]) = pack8(f0, f1);
  }
  __syncthreads();

  for (int kb = 0; kb < 96; ++kb) {
    const int kg = kb * 32;
    const bool more = (kb + 1) < 96;

    // Issue next tile's gather loads early (overlap with WMMA below)
    f32x4 nf0 = {}, nf1 = {};
    if (more) {
      gfp qn = stage_ptr(kb + 1);
      nf0 = *(gf4p)(qn);
      nf1 = *(gf4p)(qn + 4);
      __builtin_prefetch(qn + 64, 0, 1);  // two tiles ahead -> global_prefetch_b8
    }

    // B fragments (32x16 bf16, pre-transposed weights: lane = column)
    FragAB b0, b1f;
    {
      gu16p qb0 = (gu16p)(Wt + (size_t)(wave * 32 + lm) * 3072 + kg + hsel * 8);
      b0.u[0] = *(gu4p)(qb0);
      b0.u[1] = *(gu4p)(qb0 + 16);
      gu16p qb1 = qb0 + (size_t)16 * 3072;
      b1f.u[0] = *(gu4p)(qb1);
      b1f.u[1] = *(gu4p)(qb1 + 16);
    }

    const unsigned short* Ab = &As[(kb & 1) * ABUF];
#pragma unroll
    for (int mt = 0; mt < 4; ++mt) {
      const unsigned short* qa = Ab + (mt * 16 + lm) * LDA + hsel * 8;
      FragAB a;
      a.u[0] = *(const u32x4*)(qa);         // ds_load_b128
      a.u[1] = *(const u32x4*)(qa + 16);
      acc[mt][0] = __builtin_amdgcn_wmma_f32_16x16x32_bf16(
          false, a.v, false, b0.v, (short)0, acc[mt][0], false, false);
      acc[mt][1] = __builtin_amdgcn_wmma_f32_16x16x32_bf16(
          false, a.v, false, b1f.v, (short)0, acc[mt][1], false, false);
    }

    // Store next tile into the other buffer (safe: nobody reads it this step)
    if (more) {
      *(u32x4*)(&As[((kb + 1) & 1) * ABUF + srow * LDA + spart * 8]) =
          pack8(nf0, nf1);
    }
    __syncthreads();
  }

  // Epilogue: bias + ReLU. C layout: VGPR r = row r (lanes 0-15) / r+8 (16-31)
#pragma unroll
  for (int nt = 0; nt < 2; ++nt) {
    const int col = wave * 32 + nt * 16 + lm;
    const float bv = bias[col];
#pragma unroll
    for (int mt = 0; mt < 4; ++mt) {
      const long row0 = tileM + mt * 16 + hsel * 8;
#pragma unroll
      for (int r = 0; r < 8; ++r) {
        float v = acc[mt][nt][r] + bv;
        v = v > 0.0f ? v : 0.0f;
        out[(row0 + r) * 256 + col] = v;
      }
    }
  }
}

// ---------------------------------------------------------------------------
// GEMM 2: e_f_lang = relu([w2v[src] | w2v[dst]] @ Wl + bl), K=600 padded to 608
// ---------------------------------------------------------------------------
__global__ __launch_bounds__(256) void edge_fc_lang(
    const float* __restrict__ w2v,
    const unsigned short* __restrict__ Wt,  // bf16 [256][608]
    const float* __restrict__ bias,
    const int* __restrict__ src,
    const int* __restrict__ dst,
    float* __restrict__ out) {
  __shared__ __align__(128) unsigned short As[2 * ABUF];

  const int tid  = threadIdx.x;
  const int lane = tid & 31;
  const int wave = tid >> 5;
  const int hsel = lane >> 4;
  const int lm   = lane & 15;
  const long tileM = (long)blockIdx.x * 64;

  const int srow  = tid >> 2;
  const int spart = tid & 3;
  const long e = tileM + srow;
  gfp ps = (gfp)(w2v + (long)src[e] * 300);
  gfp pd = (gfp)(w2v + (long)dst[e] * 300);

  // Elementwise gather for K-tile kb (src cols 0..299 | dst cols 300..599 | pad)
  auto stage_vals = [&](int kb, f32x4& f0, f32x4& f1) {
    float f[8];
#pragma unroll
    for (int j = 0; j < 8; ++j) {
      const int kk = kb * 32 + spart * 8 + j;
      float v = 0.0f;
      if (kk < 300)      v = ps[kk];
      else if (kk < 600) v = pd[kk - 300];
      f[j] = v;
    }
    f0.x = f[0]; f0.y = f[1]; f0.z = f[2]; f0.w = f[3];
    f1.x = f[4]; f1.y = f[5]; f1.z = f[6]; f1.w = f[7];
  };

  v8f acc[4][2] = {};

  {
    f32x4 f0, f1;
    stage_vals(0, f0, f1);
    *(u32x4*)(&As[srow * LDA + spart * 8]) = pack8(f0, f1);
  }
  __syncthreads();

  for (int kb = 0; kb < 19; ++kb) {
    const int kg = kb * 32;
    const bool more = (kb + 1) < 19;

    f32x4 nf0 = {}, nf1 = {};
    if (more) stage_vals(kb + 1, nf0, nf1);

    FragAB b0, b1f;
    {
      gu16p qb0 = (gu16p)(Wt + (size_t)(wave * 32 + lm) * 608 + kg + hsel * 8);
      b0.u[0] = *(gu4p)(qb0);
      b0.u[1] = *(gu4p)(qb0 + 16);
      gu16p qb1 = qb0 + (size_t)16 * 608;
      b1f.u[0] = *(gu4p)(qb1);
      b1f.u[1] = *(gu4p)(qb1 + 16);
    }

    const unsigned short* Ab = &As[(kb & 1) * ABUF];
#pragma unroll
    for (int mt = 0; mt < 4; ++mt) {
      const unsigned short* qa = Ab + (mt * 16 + lm) * LDA + hsel * 8;
      FragAB a;
      a.u[0] = *(const u32x4*)(qa);
      a.u[1] = *(const u32x4*)(qa + 16);
      acc[mt][0] = __builtin_amdgcn_wmma_f32_16x16x32_bf16(
          false, a.v, false, b0.v, (short)0, acc[mt][0], false, false);
      acc[mt][1] = __builtin_amdgcn_wmma_f32_16x16x32_bf16(
          false, a.v, false, b1f.v, (short)0, acc[mt][1], false, false);
    }

    if (more) {
      *(u32x4*)(&As[((kb + 1) & 1) * ABUF + srow * LDA + spart * 8]) =
          pack8(nf0, nf1);
    }
    __syncthreads();
  }

#pragma unroll
  for (int nt = 0; nt < 2; ++nt) {
    const int col = wave * 32 + nt * 16 + lm;
    const float bv = bias[col];
#pragma unroll
    for (int mt = 0; mt < 4; ++mt) {
      const long row0 = tileM + mt * 16 + hsel * 8;
#pragma unroll
      for (int r = 0; r < 8; ++r) {
        float v = acc[mt][nt][r] + bv;
        v = v > 0.0f ? v : 0.0f;
        out[(row0 + r) * 256 + col] = v;
      }
    }
  }
}

// ---------------------------------------------------------------------------
// Launch
// ---------------------------------------------------------------------------
extern "C" void kernel_launch(void* const* d_in, const int* in_sizes, int n_in,
                              void* d_out, int out_size, void* d_ws, size_t ws_size,
                              hipStream_t stream) {
  const float* n_f = (const float*)d_in[0];
  const float* w2v = (const float*)d_in[1];
  const float* s_f = (const float*)d_in[2];
  const float* W1  = (const float*)d_in[3];
  const float* b1  = (const float*)d_in[4];
  const float* Wl  = (const float*)d_in[5];
  const float* bl  = (const float*)d_in[6];
  const int*   src = (const int*)d_in[7];
  const int*   dst = (const int*)d_in[8];

  const int E = in_sizes[7];  // 131072 edges

  // Workspace: bf16 transposed weights
  unsigned short* Wt1 = (unsigned short*)d_ws;                 // 256*3072*2 B
  unsigned short* Wtl = Wt1 + (size_t)256 * 3072;              // 256*608*2 B

  float* out1 = (float*)d_out;                 // e_f      [E,256]
  float* out2 = out1 + (size_t)E * 256;        // e_f_lang [E,256]

  prep_w_main<<<3072, 256, 0, stream>>>(W1, Wt1);
  prep_w_lang<<<608, 256, 0, stream>>>(Wl, Wtl);
  edge_fc_main<<<E / 64, 256, 0, stream>>>(n_f, s_f, Wt1, b1, src, dst, out1);
  edge_fc_lang<<<E / 64, 256, 0, stream>>>(w2v, Wtl, bl, src, dst, out2);
}
